// UpdateONNX_31920196943973
// MI455X (gfx1250) — compile-verified, matrix operand order checked by reading, not run on hardware
//
#include <hip/hip_runtime.h>
#include <hip/hip_bf16.h>
#include <math.h>

#define DIM 384
#define LN_EPS 1e-3f
#define MAXN 4096

typedef __bf16 bf16_t;
typedef __attribute__((ext_vector_type(16))) __bf16 bf16x16;
typedef __attribute__((ext_vector_type(8)))  __bf16 bf16x8;
typedef __attribute__((ext_vector_type(8)))  float  f32x8;
typedef __attribute__((ext_vector_type(4)))  float  f32x4;
typedef __attribute__((ext_vector_type(4)))  unsigned int u32x4;
typedef __attribute__((ext_vector_type(4)))  int    i32x4;
typedef __attribute__((ext_vector_type(8)))  int    i32x8;

// ---------------------------------------------------------------------------
// Tensor Data Mover: 2D tile load Global -> LDS (D# per CDNA5 ISA)
//   group0: [0]=count=1 (user mode), [1]=lds byte addr,
//           [2]=global_addr[31:0], [3]=global_addr[56:32] | type=2<<30
//   group1: [0]=workgroup_mask=0 | data_size<<16 | pad ctl,
//           [1..3]=tensor_dim0/1 + tile_dim0, [4]=tile_dim1,
//           [5..6]=tensor_dim0_stride (48b), [7]=0
// OOB reads return zero (handles the K=882 tail); pad_interval/amount insert
// 16B of LDS padding every 128B row -> padded f32 row stride of 36.
// ---------------------------------------------------------------------------
__device__ inline void tdm_load_2d(unsigned int lds_off, const void* gptr,
                                   unsigned int tensor_d0, unsigned int tensor_d1,
                                   unsigned int tile_d0, unsigned int tile_d1,
                                   unsigned int stride_d0, unsigned int flags_w0) {
  unsigned long long ga = (unsigned long long)gptr;
  u32x4 g0;
  g0[0] = 1u;                                         // count=1, user descriptor
  g0[1] = lds_off;                                    // lds_addr (bytes)
  g0[2] = (unsigned int)(ga & 0xFFFFFFFFu);           // global_addr[31:0]
  g0[3] = (unsigned int)((ga >> 32) & 0x1FFFFFFu) | (2u << 30);  // [56:32] | type=2
  i32x8 g1;
  g1[0] = (int)flags_w0;                              // mask=0 | data_size | pad ctl
  g1[1] = (int)((tensor_d0 & 0xFFFFu) << 16);         // atomic_addr=0 | td0[15:0]
  g1[2] = (int)((tensor_d0 >> 16) | ((tensor_d1 & 0xFFFFu) << 16));
  g1[3] = (int)((tensor_d1 >> 16) | ((tile_d0 & 0xFFFFu) << 16));
  g1[4] = (int)(tile_d1 & 0xFFFFu);                   // tile_dim1 | tile_dim2=0
  g1[5] = (int)stride_d0;                             // tensor_dim0_stride[31:0]
  g1[6] = 0;                                          // stride[47:32] | d1stride lo
  g1[7] = 0;
  i32x4 z4 = {0, 0, 0, 0};
#if __clang_major__ >= 23
  i32x8 z8 = {0, 0, 0, 0, 0, 0, 0, 0};
  __builtin_amdgcn_tensor_load_to_lds(g0, g1, z4, z4, z8, 0);
#else
  __builtin_amdgcn_tensor_load_to_lds(g0, g1, z4, z4, 0);
#endif
}

// ---------------------------------------------------------------------------
// WMMA GEMM: Y[n,O] = act(X[n,K] @ W[K,O] + bias)   (optionally Y += ...)
// 128x64 tile / 256-thread block (8 waves, 32x32 sub-block each).
// A tile: TDM -> LDS fp32 (padded stride 36), bf16-converted at fragment load.
// B tile: manual transposed fp32->bf16 staging (stride 40).
// ---------------------------------------------------------------------------
#define BM 128
#define BN 64
#define BK 32
#define LDAF 36  // f32 stride: 32 + 4 pad (TDM pad_interval=128B, amount=16B)
#define LDB  40  // bf16 stride (80B, keeps 16B alignment, spreads banks)
#define A_TDM_FLAGS ((2u << 16) | (1u << 20) | (4u << 22) | (3u << 25))

__device__ inline bf16x16 load_frag_a(const float* base) {
  // A 16x32 16-bit layout: lanes 0-15 K 0..7/16..23, lanes 16-31 K 8..15/24..31
  const int lane = threadIdx.x & 31;
  const int half = lane >> 4;
  const float* p = base + (lane & 15) * LDAF + half * 8;
  f32x4 c0 = *(const f32x4*)(p);
  f32x4 c1 = *(const f32x4*)(p + 4);
  f32x4 c2 = *(const f32x4*)(p + 16);
  f32x4 c3 = *(const f32x4*)(p + 20);
  bf16x16 r;
#pragma unroll
  for (int i = 0; i < 4; ++i) {
    r[i]      = (bf16_t)c0[i];
    r[4 + i]  = (bf16_t)c1[i];
    r[8 + i]  = (bf16_t)c2[i];
    r[12 + i] = (bf16_t)c3[i];
  }
  return r;
}

__device__ inline bf16x16 load_frag_b(const bf16_t* base) {
  const int lane = threadIdx.x & 31;
  const int half = lane >> 4;
  const bf16_t* p = base + (lane & 15) * LDB + half * 8;
  union { bf16x16 v; bf16x8 h[2]; } u;
  u.h[0] = *(const bf16x8*)(p);        // ds_load_b128
  u.h[1] = *(const bf16x8*)(p + 16);   // ds_load_b128
  return u.v;
}

template <int ACT, bool ACCUM>
__global__ __launch_bounds__(256) void gemm_kernel(
    const float* __restrict__ X, const float* __restrict__ W,
    const float* __restrict__ bias, float* __restrict__ Y,
    int Nrows, int K, int O) {
  __shared__ __align__(16) float  As[BM * LDAF];
  __shared__ __align__(16) bf16_t Bs[BN * LDB];

  const int tid     = threadIdx.x;
  const int rowBase = blockIdx.x * BM;
  const int colBase = blockIdx.y * BN;
  const int wid     = tid >> 5;
  const int lane    = tid & 31;
  const int m0      = (wid & 3) * 32;   // 4 wave-rows
  const int n0      = (wid >> 2) * 32;  // 2 wave-cols
  const unsigned int As_off = (unsigned int)(unsigned long long)(&As[0]);

  f32x8 acc00 = {}, acc01 = {}, acc10 = {}, acc11 = {};

  const int ksteps = (K + BK - 1) / BK;
  for (int kt = 0; kt < ksteps; ++kt) {
    const int k0 = kt * BK;
    // ---- A tile via Tensor Data Mover (OOB zero-fill covers K tail) ----
    if (wid == 0) {
      const float* gA = X + (size_t)rowBase * K + k0;
      tdm_load_2d(As_off, gA, (unsigned int)(K - k0),
                  (unsigned int)(Nrows - rowBase), BK, BM,
                  (unsigned int)K, A_TDM_FLAGS);
    }
    // ---- B tile: stage transposed, fp32 -> bf16:  Bs[col][k] ----
    for (int i = tid; i < BK * BN; i += 256) {
      int c = i & 63, k = i >> 6;
      int gk = k0 + k, gc = colBase + c;
      float v = (gk < K) ? W[(size_t)gk * O + gc] : 0.f;
      Bs[c * LDB + k] = (bf16_t)v;
    }
    if (wid == 0) __builtin_amdgcn_s_wait_tensorcnt(0);
    __syncthreads();

    bf16x16 a0 = load_frag_a(As + (m0 + 0)  * LDAF);
    bf16x16 a1 = load_frag_a(As + (m0 + 16) * LDAF);
    bf16x16 b0 = load_frag_b(Bs + (n0 + 0)  * LDB);
    bf16x16 b1 = load_frag_b(Bs + (n0 + 16) * LDB);

    acc00 = __builtin_amdgcn_wmma_f32_16x16x32_bf16(false, a0, false, b0, (short)0, acc00, false, false);
    acc01 = __builtin_amdgcn_wmma_f32_16x16x32_bf16(false, a0, false, b1, (short)0, acc01, false, false);
    acc10 = __builtin_amdgcn_wmma_f32_16x16x32_bf16(false, a1, false, b0, (short)0, acc10, false, false);
    acc11 = __builtin_amdgcn_wmma_f32_16x16x32_bf16(false, a1, false, b1, (short)0, acc11, false, false);

    __syncthreads();
  }

  // ---- epilogue: C/D layout -> lane (l&15)=col, VGPR r = row r (+8, l>=16)
  const int colLo = lane & 15;
  const int rOff  = (lane >> 4) << 3;
#pragma unroll
  for (int mt = 0; mt < 2; ++mt) {
#pragma unroll
    for (int nt = 0; nt < 2; ++nt) {
      f32x8 acc = (mt == 0) ? (nt == 0 ? acc00 : acc01)
                            : (nt == 0 ? acc10 : acc11);
      int col  = colBase + n0 + nt * 16 + colLo;
      int row0 = rowBase + m0 + mt * 16 + rOff;
      float bv = bias ? bias[col] : 0.f;
#pragma unroll
      for (int r = 0; r < 8; ++r) {
        int row = row0 + r;
        if (row < Nrows) {
          float v = acc[r] + bv;
          if (ACT == 1) v = fmaxf(v, 0.f);
          else if (ACT == 2) v = 1.f / (1.f + __expf(-v));
          else if (ACT == 3) v = __expf(v);
          size_t o = (size_t)row * O + col;
          if (ACCUM) Y[o] += v; else Y[o] = v;
        }
      }
    }
  }
}

// ---------------------------------------------------------------------------
// Transpose [C][N] -> [N][C] via LDS tile
// ---------------------------------------------------------------------------
__global__ void transpose_kernel(const float* __restrict__ src,
                                 float* __restrict__ dst, int C, int N) {
  __shared__ float tile[32][33];
  int nb = blockIdx.x * 32, cb = blockIdx.y * 32;
  int tx = threadIdx.x, ty = threadIdx.y;
  for (int i = ty; i < 32; i += 8) {
    int c = cb + i, n = nb + tx;
    tile[i][tx] = (c < C && n < N) ? src[(size_t)c * N + n] : 0.f;
  }
  __syncthreads();
  for (int i = ty; i < 32; i += 8) {
    int n = nb + i, c = cb + tx;
    if (n < N && c < C) dst[(size_t)n * C + c] = tile[tx][i];
  }
}

// ---------------------------------------------------------------------------
// LayerNorm over DIM (one block of 128 threads per row)
// ---------------------------------------------------------------------------
template <int ACT>
__global__ __launch_bounds__(128) void ln_kernel(
    const float* __restrict__ X, const float* __restrict__ g,
    const float* __restrict__ b, float* __restrict__ Y) {
  __shared__ float h[DIM];
  __shared__ float red[128];
  int row = blockIdx.x, tid = threadIdx.x;
  float s = 0.f;
  for (int c = tid; c < DIM; c += 128) { float v = X[(size_t)row * DIM + c]; h[c] = v; s += v; }
  red[tid] = s; __syncthreads();
  for (int o = 64; o > 0; o >>= 1) { if (tid < o) red[tid] += red[tid + o]; __syncthreads(); }
  float m = red[0] * (1.f / DIM);
  __syncthreads();
  float s2 = 0.f;
  for (int c = tid; c < DIM; c += 128) { float d = h[c] - m; s2 += d * d; }
  red[tid] = s2; __syncthreads();
  for (int o = 64; o > 0; o >>= 1) { if (tid < o) red[tid] += red[tid + o]; __syncthreads(); }
  float rstd = rsqrtf(red[0] * (1.f / DIM) + LN_EPS);
  for (int c = tid; c < DIM; c += 128) {
    float v = (h[c] - m) * rstd * g[c] + b[c];
    if (ACT == 1) v = fmaxf(v, 0.f);
    Y[(size_t)row * DIM + c] = v;
  }
}

// net = LN(net_cm + inp_cm + c_rm)  (net/inp are channel-major [C][N])
__global__ __launch_bounds__(128) void add3_ln_kernel(
    const float* __restrict__ net_cm, const float* __restrict__ inp_cm,
    const float* __restrict__ c_rm, const float* __restrict__ g,
    const float* __restrict__ b, float* __restrict__ out_rm, int N) {
  __shared__ float h[DIM];
  __shared__ float red[128];
  int row = blockIdx.x, tid = threadIdx.x;
  float s = 0.f;
  for (int c = tid; c < DIM; c += 128) {
    float v = net_cm[(size_t)c * N + row] + inp_cm[(size_t)c * N + row] +
              c_rm[(size_t)row * DIM + c];
    h[c] = v; s += v;
  }
  red[tid] = s; __syncthreads();
  for (int o = 64; o > 0; o >>= 1) { if (tid < o) red[tid] += red[tid + o]; __syncthreads(); }
  float m = red[0] * (1.f / DIM);
  __syncthreads();
  float s2 = 0.f;
  for (int c = tid; c < DIM; c += 128) { float d = h[c] - m; s2 += d * d; }
  red[tid] = s2; __syncthreads();
  for (int o = 64; o > 0; o >>= 1) { if (tid < o) red[tid] += red[tid + o]; __syncthreads(); }
  float rstd = rsqrtf(red[0] * (1.f / DIM) + LN_EPS);
  for (int c = tid; c < DIM; c += 128)
    out_rm[(size_t)row * DIM + c] = (h[c] - m) * rstd * g[c] + b[c];
}

// ---------------------------------------------------------------------------
// Neighbor search + first-occurrence group ids (ii/jj/kk cached in LDS)
// ---------------------------------------------------------------------------
__global__ __launch_bounds__(256) void index_prep_kernel(
    const int* __restrict__ ii, const int* __restrict__ jj,
    const int* __restrict__ kk, int* __restrict__ ix, int* __restrict__ jx,
    int* __restrict__ idxk, int* __restrict__ idxij, int N) {
  __shared__ int sii[MAXN], sjj[MAXN], skk[MAXN];
  int tid = threadIdx.x;
  for (int i = tid; i < N; i += 256) { sii[i] = ii[i]; sjj[i] = jj[i]; skk[i] = kk[i]; }
  __syncthreads();
  int n = blockIdx.x * 256 + tid;
  if (n >= N) return;
  int kn = skk[n], jn = sjj[n];
  int keyn = sii[n] * 12345 + jn;
  int bestP = -1, bi = 0;     // argmax of (jm if same-k & jm<jn else 0), first max
  int bestM = N + 1, bj = 0;  // argmin of (jm if same-k & jm>jn else N), first min
  int fk = -1, fij = -1;      // first occurrence group ids
  for (int m = 0; m < N; ++m) {
    int km = skk[m], jm = sjj[m];
    int pv = (km == kn && jm < jn) ? jm : 0;
    if (pv > bestP) { bestP = pv; bi = m; }
    int nv = (km == kn && jm > jn) ? jm : N;
    if (nv < bestM) { bestM = nv; bj = m; }
    if (fk < 0 && km == kn) fk = m;
    if (fij < 0 && sii[m] * 12345 + jm == keyn) fij = m;
  }
  ix[n] = bi; jx[n] = bj; idxk[n] = fk; idxij[n] = fij;
}

// ---------------------------------------------------------------------------
// Elementwise / segment helpers
// ---------------------------------------------------------------------------
__global__ void zero_kernel(float* p, size_t n) {
  size_t i = (size_t)blockIdx.x * 256 + threadIdx.x;
  if (i < n) p[i] = 0.f;
}

__global__ void seg_accum_kernel(const float* __restrict__ fx,
                                 const float* __restrict__ ex,
                                 const int* __restrict__ idx,
                                 float* __restrict__ num,
                                 float* __restrict__ den, int N) {
  size_t i = (size_t)blockIdx.x * 256 + threadIdx.x;
  if (i >= (size_t)N * DIM) return;
  int n = (int)(i / DIM), c = (int)(i % DIM);
  float e = ex[i];
  size_t o = (size_t)idx[n] * DIM + c;
  atomicAdd(&num[o], fx[i] * e);
  atomicAdd(&den[o], e);
}

__global__ void div_kernel(float* __restrict__ num, const float* __restrict__ den, size_t n) {
  size_t i = (size_t)blockIdx.x * 256 + threadIdx.x;
  if (i < n) { float d = den[i]; num[i] = (d != 0.f) ? num[i] / d : 0.f; }
}

__global__ void gather_rows_kernel(const float* __restrict__ src,
                                   const int* __restrict__ idx,
                                   float* __restrict__ dst, int N) {
  size_t i = (size_t)blockIdx.x * 256 + threadIdx.x;
  if (i >= (size_t)N * DIM) return;
  int n = (int)(i / DIM), c = (int)(i % DIM);
  dst[i] = src[(size_t)idx[n] * DIM + c];
}

__global__ void gather_add_kernel(float* __restrict__ dst,
                                  const float* __restrict__ src,
                                  const int* __restrict__ idx, int N) {
  size_t i = (size_t)blockIdx.x * 256 + threadIdx.x;
  if (i >= (size_t)N * DIM) return;
  int n = (int)(i / DIM), c = (int)(i % DIM);
  dst[i] += src[(size_t)idx[n] * DIM + c];
}

__global__ void gated_add_kernel(const float* __restrict__ x, const float* __restrict__ g,
                                 const float* __restrict__ r, float* __restrict__ out, size_t n) {
  size_t i = (size_t)blockIdx.x * 256 + threadIdx.x;
  if (i < n) out[i] = x[i] + g[i] * r[i];
}

// ---------------------------------------------------------------------------
// Output heads: d = relu(net)@dw+db ; w = sigmoid(relu(net)@ww+wb)  (O=2)
// One wave32 per row.
// ---------------------------------------------------------------------------
__global__ __launch_bounds__(128) void head_kernel(
    const float* __restrict__ net, const float* __restrict__ dw,
    const float* __restrict__ db, const float* __restrict__ ww,
    const float* __restrict__ wb, float* __restrict__ dout,
    float* __restrict__ wout, int N) {
  int warp = (blockIdx.x * blockDim.x + threadIdx.x) >> 5;
  int lane = threadIdx.x & 31;
  if (warp >= N) return;
  const float* x = net + (size_t)warp * DIM;
  float s00 = 0.f, s01 = 0.f, s10 = 0.f, s11 = 0.f;
  for (int k = lane; k < DIM; k += 32) {
    float v = fmaxf(x[k], 0.f);
    s00 += v * dw[k * 2 + 0]; s01 += v * dw[k * 2 + 1];
    s10 += v * ww[k * 2 + 0]; s11 += v * ww[k * 2 + 1];
  }
  for (int off = 16; off > 0; off >>= 1) {
    s00 += __shfl_down(s00, off, 32);
    s01 += __shfl_down(s01, off, 32);
    s10 += __shfl_down(s10, off, 32);
    s11 += __shfl_down(s11, off, 32);
  }
  if (lane == 0) {
    dout[warp * 2 + 0] = s00 + db[0];
    dout[warp * 2 + 1] = s01 + db[1];
    wout[warp * 2 + 0] = 1.f / (1.f + __expf(-(s10 + wb[0])));
    wout[warp * 2 + 1] = 1.f / (1.f + __expf(-(s11 + wb[1])));
  }
}

// ---------------------------------------------------------------------------
// Host orchestration
// ---------------------------------------------------------------------------
static void gemm(int act, bool accum, const float* X, const float* W, const float* bias,
                 float* Y, int n, int K, int O, hipStream_t s) {
  dim3 grid((n + BM - 1) / BM, (O + BN - 1) / BN), blk(256);
  if (accum) { gemm_kernel<0, true><<<grid, blk, 0, s>>>(X, W, bias, Y, n, K, O); return; }
  switch (act) {
    case 1:  gemm_kernel<1, false><<<grid, blk, 0, s>>>(X, W, bias, Y, n, K, O); break;
    case 2:  gemm_kernel<2, false><<<grid, blk, 0, s>>>(X, W, bias, Y, n, K, O); break;
    case 3:  gemm_kernel<3, false><<<grid, blk, 0, s>>>(X, W, bias, Y, n, K, O); break;
    default: gemm_kernel<0, false><<<grid, blk, 0, s>>>(X, W, bias, Y, n, K, O); break;
  }
}

extern "C" void kernel_launch(void* const* d_in, const int* in_sizes, int n_in,
                              void* d_out, int out_size, void* d_ws, size_t ws_size,
                              hipStream_t stream) {
  const float* net_cm  = (const float*)d_in[0];
  const float* inp_cm  = (const float*)d_in[1];
  const float* corr_cm = (const float*)d_in[2];
  const int*   ii      = (const int*)d_in[3];
  const int*   jj      = (const int*)d_in[4];
  const int*   kk      = (const int*)d_in[5];
  const int N       = in_sizes[3];           // 4096
  const int CORR_IN = in_sizes[2] / N;       // 882
  auto P = [&](int k) { return (const float*)d_in[6 + k]; };
  // Param order (make_params dict insertion order):
  // 0 corr1w 1 corr1b 2 corr2w 3 corr2b 4 corrlng 5 corrlnb 6 corr3w 7 corr3b
  // 8 normg 9 normb 10 c1aw 11 c1ab 12 c1bw 13 c1bb 14 c2aw 15 c2ab 16 c2bw 17 c2bb
  // 18 akfw 19 akfb 20 akgw 21 akgb 22 akhw 23 akhb 24 aifw 25 aifb 26 aigw 27 aigb
  // 28 aihw 29 aihb 30 gln1g 31 gln1b 32 gr1gw 33 gr1gb 34 gr1aw 35 gr1ab 36 gr1bw
  // 37 gr1bb 38 gln2g 39 gln2b 40 gr2gw 41 gr2gb 42 gr2aw 43 gr2ab 44 gr2bw 45 gr2bb
  // 46 dw 47 db 48 ww 49 wb

  const size_t NF = (size_t)N * DIM;
  float* ws      = (float*)d_ws;
  float* corr_rm = ws;                                 // N * CORR_IN
  float* net_rm  = corr_rm + (size_t)N * CORR_IN;      // NF
  float* x_rm    = net_rm + NF;                        // NF
  float* b1      = x_rm + NF;                          // NF
  float* b2      = b1 + NF;                            // NF
  float* b3      = b2 + NF;                            // NF
  float* b4      = b3 + NF;                            // NF
  int*   ixb     = (int*)(b4 + NF);
  int*   jxb     = ixb + N;
  int*   idxk    = jxb + N;
  int*   idxij   = idxk + N;

  const int EW = (int)((NF + 255) / 256);

  // 1) transpose corr to row-major; compute neighbor/group indices
  {
    dim3 gT((N + 31) / 32, (CORR_IN + 31) / 32), bT(32, 8);
    transpose_kernel<<<gT, bT, 0, stream>>>(corr_cm, corr_rm, CORR_IN, N);
  }
  index_prep_kernel<<<(N + 255) / 256, 256, 0, stream>>>(ii, jj, kk, ixb, jxb, idxk, idxij, N);

  // 2) corr MLP: relu(L1) -> L2 -> relu(LN) -> L3
  gemm(1, false, corr_rm, P(0), P(1), b1, N, CORR_IN, DIM, stream);
  gemm(0, false, b1, P(2), P(3), b2, N, DIM, DIM, stream);
  ln_kernel<1><<<N, 128, 0, stream>>>(b2, P(4), P(5), b1);
  gemm(0, false, b1, P(6), P(7), b2, N, DIM, DIM, stream);   // c in b2

  // 3) net = LN(net + inp + c)
  add3_ln_kernel<<<N, 128, 0, stream>>>(net_cm, inp_cm, b2, P(8), P(9), net_rm, N);

  // 4) net += mlp2(net[ix]) ; net += mlp2(net[jx])
  gather_rows_kernel<<<EW, 256, 0, stream>>>(net_rm, ixb, b1, N);
  gemm(1, false, b1, P(10), P(11), b2, N, DIM, DIM, stream);
  gemm(0, true,  b2, P(12), P(13), net_rm, N, DIM, DIM, stream);
  gather_rows_kernel<<<EW, 256, 0, stream>>>(net_rm, jxb, b1, N);
  gemm(1, false, b1, P(14), P(15), b2, N, DIM, DIM, stream);
  gemm(0, true,  b2, P(16), P(17), net_rm, N, DIM, DIM, stream);

  // 5) segment-softmax aggregations (kk groups, then (ii,jj) groups)
  for (int pass = 0; pass < 2; ++pass) {
    const int*   idx = pass == 0 ? idxk : idxij;
    const float *fw = P(pass ? 24 : 18), *fb = P(pass ? 25 : 19);
    const float *gw = P(pass ? 26 : 20), *gb = P(pass ? 27 : 21);
    const float *hw = P(pass ? 28 : 22), *hb = P(pass ? 29 : 23);
    gemm(0, false, net_rm, fw, fb, b1, N, DIM, DIM, stream);   // fx
    gemm(3, false, net_rm, gw, gb, b2, N, DIM, DIM, stream);   // ex = exp(gx)
    zero_kernel<<<EW, 256, 0, stream>>>(b3, NF);
    zero_kernel<<<EW, 256, 0, stream>>>(b4, NF);
    seg_accum_kernel<<<EW, 256, 0, stream>>>(b1, b2, idx, b3, b4, N);
    div_kernel<<<EW, 256, 0, stream>>>(b3, b4, NF);            // s = num/den
    gemm(0, false, b3, hw, hb, b1, N, DIM, DIM, stream);       // y per segment
    gather_add_kernel<<<EW, 256, 0, stream>>>(net_rm, b1, idx, N);
  }

  // 6) x = LN(net); x = gated_res1(x); x = LN(x); net = gated_res2(x)
  ln_kernel<0><<<N, 128, 0, stream>>>(net_rm, P(30), P(31), x_rm);
  gemm(2, false, x_rm, P(32), P(33), b1, N, DIM, DIM, stream);   // gate
  gemm(1, false, x_rm, P(34), P(35), b2, N, DIM, DIM, stream);   // relu(a)
  gemm(0, false, b2,   P(36), P(37), b3, N, DIM, DIM, stream);   // res
  gated_add_kernel<<<EW, 256, 0, stream>>>(x_rm, b1, b3, net_rm, NF);

  ln_kernel<0><<<N, 128, 0, stream>>>(net_rm, P(38), P(39), x_rm);
  gemm(2, false, x_rm, P(40), P(41), b1, N, DIM, DIM, stream);
  gemm(1, false, x_rm, P(42), P(43), b2, N, DIM, DIM, stream);
  gemm(0, false, b2,   P(44), P(45), b3, N, DIM, DIM, stream);
  gated_add_kernel<<<EW, 256, 0, stream>>>(x_rm, b1, b3, net_rm, NF);

  // 7) outputs: [net | d_out | w_out] flat
  float* out = (float*)d_out;
  hipMemcpyAsync(out, net_rm, NF * sizeof(float), hipMemcpyDeviceToDevice, stream);
  head_kernel<<<(N * 32 + 127) / 128, 128, 0, stream>>>(
      net_rm, P(46), P(47), P(48), P(49), out + NF, out + NF + (size_t)2 * N, N);
}
